// EnhancedFeatureEncoder_62371515072987
// MI455X (gfx1250) — compile-verified, matrix operand order checked by reading, not run on hardware
//
#include <hip/hip_runtime.h>
#include <hip/hip_bf16.h>
#include <stdint.h>

typedef __attribute__((ext_vector_type(16))) __bf16 v16bf;
typedef __attribute__((ext_vector_type(8)))  float  v8f;

#define LN_EPS 1e-5f

__device__ __forceinline__ unsigned short f2bf(float f) {
  __bf16 b = (__bf16)f;
  return __builtin_bit_cast(unsigned short, b);
}
__device__ __forceinline__ float bf2f(unsigned short u) {
  unsigned x = ((unsigned)u) << 16;
  return __builtin_bit_cast(float, x);
}
__device__ __forceinline__ v8f zero8() { v8f z = {0,0,0,0,0,0,0,0}; return z; }

// LayerNorm across one wave32: each lane holds CPL consecutive channels of an
// n-channel vector. Reduction via lane shuffles (wave32-correct).
template<int CPL>
__device__ __forceinline__ void lane_ln(float (&v)[CPL], float n) {
  float s = 0.f;
#pragma unroll
  for (int i = 0; i < CPL; ++i) s += v[i];
#pragma unroll
  for (int off = 16; off; off >>= 1) s += __shfl_xor(s, off, 32);
  float mu = s / n;
  float s2 = 0.f;
#pragma unroll
  for (int i = 0; i < CPL; ++i) { float d = v[i] - mu; s2 += d * d; }
#pragma unroll
  for (int off = 16; off; off >>= 1) s2 += __shfl_xor(s2, off, 32);
  float r = rsqrtf(s2 / n + LN_EPS);
#pragma unroll
  for (int i = 0; i < CPL; ++i) v[i] = (v[i] - mu) * r;
}

// CDNA5 async global -> LDS copy (ASYNCcnt path). Per-lane: 16B from
// (SADDR base + 32-bit VGPR byte offset) into LDS at the lane's VDST offset.
__device__ __forceinline__ void async_b128(void* lds_dst, const void* gbase,
                                           unsigned byte_off) {
  unsigned lds_off = (unsigned)(unsigned long long)lds_dst;  // low 32b of flat LDS addr == LDS offset
  asm volatile("global_load_async_to_lds_b128 %0, %1, %2"
               :: "v"(lds_off), "v"(byte_off), "s"(gbase)
               : "memory");
}
__device__ __forceinline__ void wait_async() {
  asm volatile("s_wait_asynccnt 0" ::: "memory");
}

// Stage a 32(K) x 256(N) f32 weight panel (32KB) into LDS: 512 threads x 16 floats.
__device__ __forceinline__ void stage_panel(float* sBuf, const float* gsrc, int tid) {
  int e0 = tid * 16;
#pragma unroll
  for (int i = 0; i < 4; ++i)
    async_b128(&sBuf[e0 + i * 4], gsrc, (unsigned)((e0 + i * 4) * sizeof(float)));
  wait_async();
}

// One wave computes D[16 x 32] (two 16x16 N-tiles) of A[Mtiles*16 x K] @ W[K x 256],
// with W streamed through sBuf in 32-K panels. A is bf16 in LDS (row stride astride).
__device__ __forceinline__ void gemm2(const unsigned short* sA, int astride,
                                      const float* W, int K, float* sBuf,
                                      int tid, int m, int half, int nl, int nt0,
                                      v8f& acc0, v8f& acc1) {
  for (int ks = 0; ks < K; ks += 32) {
    stage_panel(sBuf, W + (size_t)ks * 256, tid);
    __syncthreads();
    // A fragment: 16-bit A 16x32 layout -> lane half owns K {0-7,16-23} / {8-15,24-31}
    union { v16bf v; unsigned u[8]; } a;
#pragma unroll
    for (int p = 0; p < 8; ++p) {
      int k = ks + (p >> 2) * 16 + half * 8 + 2 * (p & 3);
      a.u[p] = *(const unsigned*)&sA[m * astride + k];  // two bf16 (k, k+1)
    }
    // B fragments (32x16), f32 panel -> bf16 at load
    v16bf b0, b1;
#pragma unroll
    for (int j = 0; j < 16; ++j) {
      int k = (j >> 3) * 16 + half * 8 + (j & 7);
      b0[j] = (__bf16)sBuf[k * 256 + nt0 * 16 + nl];
      b1[j] = (__bf16)sBuf[k * 256 + nt0 * 16 + 16 + nl];
    }
    acc0 = __builtin_amdgcn_wmma_f32_16x16x32_bf16(false, a.v, false, b0,
                                                   (short)0, acc0, false, false);
    acc1 = __builtin_amdgcn_wmma_f32_16x16x32_bf16(false, a.v, false, b1,
                                                   (short)0, acc1, false, false);
    __syncthreads();  // protect sBuf before next panel overwrite
  }
}

__global__ __launch_bounds__(512)
void encoder_fused_kernel(const int* __restrict__ event_type,
                          const int* __restrict__ sku_id,
                          const int* __restrict__ url_id,
                          const int* __restrict__ cat_id,
                          const int* __restrict__ price_id,
                          const int* __restrict__ word_id,
                          const float* __restrict__ event_emb,
                          const float* __restrict__ word_emb,
                          const float* __restrict__ sku_emb,
                          const float* __restrict__ sku_W,
                          const float* __restrict__ sku_b,
                          const float* __restrict__ cat_emb,
                          const float* __restrict__ price_emb,
                          const float* __restrict__ fc1_W,
                          const float* __restrict__ fc1_b,
                          const float* __restrict__ url_emb,
                          const float* __restrict__ url_W,
                          const float* __restrict__ url_b,
                          float* __restrict__ out,
                          float* __restrict__ out_mask) {
  __shared__ unsigned short sX[32 * 1024];    // concat [se|ce|pe|we] bf16 (64KB)
  __shared__ unsigned short sAsku[32 * 128];  // LN(sku_emb) bf16
  __shared__ unsigned short sAurl[32 * 128];  // LN(url_emb) bf16
  __shared__ float sB[32 * 256];              // weight panel staging / proj tmp (32KB)
  __shared__ float sUE[32 * 256];             // url branch result f32 (32KB)
  __shared__ int sET[32];

  const int tid  = threadIdx.x;
  const int lane = tid & 31;
  const int wv   = tid >> 5;          // 16 waves
  const int base = blockIdx.x * 32;   // 32 rows per block

  // ---------------- Phase A: gathers + LayerNorms (2 rows per wave) -------------
  for (int rr = 0; rr < 2; ++rr) {
    const int r = wv * 2 + rr;
    const int g = base + r;
    const int et = event_type[g];
    if (lane == 0) { sET[r] = et; out_mask[g] = (et == 0) ? 1.0f : 0.0f; }

    // event embedding (64) -> LN -> out[:, 0:64]
    {
      float v[2];
      const float* p = event_emb + et * 64 + lane * 2;
      v[0] = p[0]; v[1] = p[1];
      lane_ln<2>(v, 64.f);
      float* o = out + (size_t)g * 320 + lane * 2;
      o[0] = v[0]; o[1] = v[1];
    }
    const int c8 = lane * 8;
    // cat (256) -> LN -> x[:, 256:512]
    {
      float v[8];
      const float* p = cat_emb + (size_t)cat_id[g] * 256 + c8;
#pragma unroll
      for (int i = 0; i < 8; ++i) v[i] = p[i];
      lane_ln<8>(v, 256.f);
#pragma unroll
      for (int i = 0; i < 8; ++i) sX[r * 1024 + 256 + c8 + i] = f2bf(v[i]);
    }
    // price (256) -> LN -> x[:, 512:768]
    {
      float v[8];
      const float* p = price_emb + (size_t)price_id[g] * 256 + c8;
#pragma unroll
      for (int i = 0; i < 8; ++i) v[i] = p[i];
      lane_ln<8>(v, 256.f);
#pragma unroll
      for (int i = 0; i < 8; ++i) sX[r * 1024 + 512 + c8 + i] = f2bf(v[i]);
    }
    // word: mean over 8 gathers (256) -> LN -> x[:, 768:1024]  (also = query branch)
    {
      float v[8] = {0, 0, 0, 0, 0, 0, 0, 0};
#pragma unroll
      for (int w = 0; w < 8; ++w) {
        const float* p = word_emb + (size_t)word_id[g * 8 + w] * 256 + c8;
#pragma unroll
        for (int i = 0; i < 8; ++i) v[i] += p[i];
      }
#pragma unroll
      for (int i = 0; i < 8; ++i) v[i] *= 0.125f;
      lane_ln<8>(v, 256.f);
#pragma unroll
      for (int i = 0; i < 8; ++i) sX[r * 1024 + 768 + c8 + i] = f2bf(v[i]);
    }
    // sku (128) -> LN -> A_sku
    {
      float v[4];
      const float* p = sku_emb + (size_t)sku_id[g] * 128 + lane * 4;
#pragma unroll
      for (int i = 0; i < 4; ++i) v[i] = p[i];
      lane_ln<4>(v, 128.f);
#pragma unroll
      for (int i = 0; i < 4; ++i) sAsku[r * 128 + lane * 4 + i] = f2bf(v[i]);
    }
    // url (128) -> LN -> A_url
    {
      float v[4];
      const float* p = url_emb + (size_t)url_id[g] * 128 + lane * 4;
#pragma unroll
      for (int i = 0; i < 4; ++i) v[i] = p[i];
      lane_ln<4>(v, 128.f);
#pragma unroll
      for (int i = 0; i < 4; ++i) sAurl[r * 128 + lane * 4 + i] = f2bf(v[i]);
    }
  }
  __syncthreads();

  // WMMA tile ownership: wave -> (M-tile, two N-tiles)
  const int m_tile = wv & 1;
  const int nt0    = (wv >> 1) * 2;
  const int nl     = lane & 15;
  const int half   = lane >> 4;
  const int m      = m_tile * 16 + nl;

  v8f acc0, acc1;

  // ---------------- Phase B: sku projection (K=128) + LN + relu ----------------
  acc0 = zero8(); acc1 = zero8();
  gemm2(sAsku, 128, sku_W, 128, sB, tid, m, half, nl, nt0, acc0, acc1);
#pragma unroll
  for (int t = 0; t < 2; ++t) {
    v8f acc = t ? acc1 : acc0;
    int ncol = (nt0 + t) * 16 + nl;
    float bb = sku_b[ncol];
#pragma unroll
    for (int rr = 0; rr < 8; ++rr)
      sB[(m_tile * 16 + half * 8 + rr) * 256 + ncol] = acc[rr] + bb;
  }
  __syncthreads();
  for (int rr = 0; rr < 2; ++rr) {   // row-wise LN then relu -> x[:, 0:256]
    int r = wv * 2 + rr;
    float v[8];
#pragma unroll
    for (int i = 0; i < 8; ++i) v[i] = sB[r * 256 + lane * 8 + i];
    lane_ln<8>(v, 256.f);
#pragma unroll
    for (int i = 0; i < 8; ++i)
      sX[r * 1024 + lane * 8 + i] = f2bf(fmaxf(v[i], 0.f));
  }
  __syncthreads();

  // ---------------- Phase C: url projection (K=128) + LN + relu ----------------
  acc0 = zero8(); acc1 = zero8();
  gemm2(sAurl, 128, url_W, 128, sB, tid, m, half, nl, nt0, acc0, acc1);
#pragma unroll
  for (int t = 0; t < 2; ++t) {
    v8f acc = t ? acc1 : acc0;
    int ncol = (nt0 + t) * 16 + nl;
    float bb = url_b[ncol];
#pragma unroll
    for (int rr = 0; rr < 8; ++rr)
      sB[(m_tile * 16 + half * 8 + rr) * 256 + ncol] = acc[rr] + bb;
  }
  __syncthreads();
  for (int rr = 0; rr < 2; ++rr) {   // LN then relu -> sUE (f32)
    int r = wv * 2 + rr;
    float v[8];
#pragma unroll
    for (int i = 0; i < 8; ++i) v[i] = sB[r * 256 + lane * 8 + i];
    lane_ln<8>(v, 256.f);
#pragma unroll
    for (int i = 0; i < 8; ++i)
      sUE[r * 256 + lane * 8 + i] = fmaxf(v[i], 0.f);
  }
  __syncthreads();

  // ---------------- Phase D: fc1 (K=1024) + relu + branch select + store -------
  acc0 = zero8(); acc1 = zero8();
  gemm2(sX, 1024, fc1_W, 1024, sB, tid, m, half, nl, nt0, acc0, acc1);

#pragma unroll
  for (int t = 0; t < 2; ++t) {
    v8f acc = t ? acc1 : acc0;
    int ncol = (nt0 + t) * 16 + nl;
    float bb = fc1_b[ncol];
#pragma unroll
    for (int rr = 0; rr < 8; ++rr) {
      int mrow = m_tile * 16 + half * 8 + rr;
      int g = base + mrow;
      int et = sET[mrow];
      float item = fmaxf(acc[rr] + bb, 0.f);
      float val = 0.f;
      if (et >= 2 && et <= 4)      val = item;                                // sku branch
      else if (et == 5)            val = sUE[mrow * 256 + ncol];              // url branch
      else if (et == 6)            val = bf2f(sX[mrow * 1024 + 768 + ncol]);  // query branch
      out[(size_t)g * 320 + 64 + ncol] = val;
    }
  }
}

extern "C" void kernel_launch(void* const* d_in, const int* in_sizes, int n_in,
                              void* d_out, int out_size, void* d_ws, size_t ws_size,
                              hipStream_t stream) {
  const int*   event_type = (const int*)d_in[0];
  const int*   sku_id     = (const int*)d_in[1];
  const int*   url_id     = (const int*)d_in[2];
  const int*   cat_id     = (const int*)d_in[3];
  const int*   price_id   = (const int*)d_in[4];
  const int*   word_id    = (const int*)d_in[5];
  const float* event_emb  = (const float*)d_in[6];
  const float* word_emb   = (const float*)d_in[7];
  const float* sku_emb    = (const float*)d_in[8];
  const float* sku_W      = (const float*)d_in[9];
  const float* sku_b      = (const float*)d_in[10];
  const float* cat_emb    = (const float*)d_in[11];
  const float* price_emb  = (const float*)d_in[12];
  const float* fc1_W      = (const float*)d_in[13];
  const float* fc1_b      = (const float*)d_in[14];
  const float* url_emb    = (const float*)d_in[15];
  const float* url_W      = (const float*)d_in[16];
  const float* url_b      = (const float*)d_in[17];

  float* out = (float*)d_out;
  const int BS = in_sizes[0];                // B*S = 51200
  float* out_mask = out + (size_t)BS * 320;  // tuple output #2

  const int blocks = BS / 32;                // 32 rows per block
  encoder_fused_kernel<<<dim3(blocks), dim3(512), 0, stream>>>(
      event_type, sku_id, url_id, cat_id, price_id, word_id,
      event_emb, word_emb, sku_emb, sku_W, sku_b, cat_emb, price_emb,
      fc1_W, fc1_b, url_emb, url_W, url_b, out, out_mask);
}
